// GINGraphPropertyModel_53291954208835
// MI455X (gfx1250) — matrix operand order repaired and checked by reading, not compiled, not run on hardware
//
#include <hip/hip_runtime.h>
#include <hip/hip_bf16.h>

#define NNODES  100000
#define NEDGES  600000
#define HDIM    128
#define NLAYERS 5
#define NGRAPH  1000
#define NFEAT   9
#define VOCAB   128
#define BNEPS   1e-5f

#define WS_STRIDE 132   // padded LDS stride for W (bf16)
#define ZT_STRIDE 136   // padded LDS stride for activation tile (bf16)

#if defined(__gfx1250__) && __has_builtin(__builtin_amdgcn_global_load_async_to_lds_b128)
#define HAVE_ASYNC_LDS 1
#else
#define HAVE_ASYNC_LDS 0
#endif

typedef __attribute__((ext_vector_type(16))) __bf16 v16bf;
typedef __attribute__((ext_vector_type(8)))  float  v8f;
typedef int v4i __attribute__((__vector_size__(4 * sizeof(int))));

__device__ __forceinline__ void atomAddF(float* p, float v) {
    __hip_atomic_fetch_add(p, v, __ATOMIC_RELAXED, __HIP_MEMORY_SCOPE_AGENT);
}

// ---------------------------------------------------------------------------
// AtomEncoder: h[n,c] = sum_f atom_emb[f][x[n,f]][c]
// ---------------------------------------------------------------------------
__global__ void atom_encoder_kernel(const int* __restrict__ x,
                                    const float* __restrict__ emb,
                                    float* __restrict__ h) {
    const int n = blockIdx.x;
    const int c = threadIdx.x;
    float acc = 0.f;
#pragma unroll
    for (int f = 0; f < NFEAT; ++f) {
        const int idx = x[n * NFEAT + f];
        acc += emb[((size_t)f * VOCAB + idx) * HDIM + c];
    }
    h[(size_t)n * HDIM + c] = acc;
}

// ---------------------------------------------------------------------------
// Edge scatter: z[dst] += h[src].  One wave per edge, float4 per lane.
// ---------------------------------------------------------------------------
__global__ __launch_bounds__(256)
void edge_scatter_kernel(const int* __restrict__ ei,
                         const float* __restrict__ h,
                         float* __restrict__ z) {
    const int wv   = threadIdx.x >> 5;
    const int lane = threadIdx.x & 31;
    const int e    = blockIdx.x * 8 + wv;
    if (e >= NEDGES) return;
    const int src = ei[e];
    const int dst = ei[NEDGES + e];
    const float4 v = *(const float4*)(h + (size_t)src * HDIM + lane * 4);
    float* zp = z + (size_t)dst * HDIM + lane * 4;
    atomAddF(zp + 0, v.x);
    atomAddF(zp + 1, v.y);
    atomAddF(zp + 2, v.z);
    atomAddF(zp + 3, v.w);
}

// ---------------------------------------------------------------------------
// Fused GEMM: out = [maybe relu(in*scaleIn+shiftIn)] @ W + bias
//             optionally accumulate per-column sum / sum-of-squares (for BN).
// 32-row macro-tiles; 8 waves/block; wave = 16-row sub-tile x 32-col pair.
// A fragment shared across the two column tiles (2 WMMA per A load).
// Macro-tile is staged into LDS with GLOBAL_LOAD_ASYNC_TO_LDS (ASYNCcnt),
// then converted f32 -> bf16 (+ fused BN/ReLU) into the padded WMMA tile.
// ---------------------------------------------------------------------------
__global__ __launch_bounds__(256)
void gemm_bn_kernel(const float* __restrict__ in,
                    const float* __restrict__ w,
                    const float* __restrict__ bias,
                    const float* __restrict__ scaleIn,   // nullable
                    const float* __restrict__ shiftIn,   // nullable
                    float* __restrict__ out,
                    float* __restrict__ colsum,          // nullable
                    float* __restrict__ colsumsq,        // nullable
                    int nrows, int macrosPerBlock, int numMacros) {
    __shared__ __bf16 Ws[HDIM * WS_STRIDE];
    __shared__ __bf16 Zt[32 * ZT_STRIDE];
    __shared__ float  Ztf[32 * HDIM];     // async staging (16 KB, contiguous)

    const int tid = threadIdx.x;

    // W (f32) -> bf16 LDS, loaded once per block
    for (int i = tid; i < HDIM * HDIM; i += 256) {
        Ws[(i >> 7) * WS_STRIDE + (i & 127)] = (__bf16)w[i];
    }

    const int lane    = tid & 31;
    const int wv      = tid >> 5;
    const int r       = lane & 15;   // A row within tile / C,D column within tile
    const int kh      = lane >> 4;   // half-select per ISA 16-bit layouts
    const int rt      = wv >> 2;               // row sub-tile (0/1)
    const int colBase = (wv & 3) * 32;         // two adjacent 16-col tiles
    const int colA    = colBase + r;
    const int colB    = colBase + 16 + r;
    const float bA    = bias[colA];
    const float bB    = bias[colB];
    const bool doScale = (scaleIn != nullptr);
    const int  arow    = rt * 16 + r;

    for (int m = 0; m < macrosPerBlock; ++m) {
        const int macro = blockIdx.x * macrosPerBlock + m;
        if (macro >= numMacros) break;
        const int row0 = macro * 32;
        const bool fullTile = (row0 + 32 <= nrows);   // block-uniform

        __syncthreads();   // protect Zt/Ztf reuse (and Ws on first iter)

        // ---- stage raw f32 macro-tile (32 x 128, contiguous 16 KB) --------
        if (fullTile) {
#if HAVE_ASYNC_LDS
            // 4 x b128 per thread, tracked on ASYNCcnt (no VGPR round-trip)
            auto g1 = (__attribute__((address_space(1))) v4i*)
                          (in + (size_t)row0 * HDIM + tid * 16);
            auto l3 = (__attribute__((address_space(3))) v4i*)
                          (Ztf + tid * 16);
            __builtin_amdgcn_global_load_async_to_lds_b128(g1, l3, 0, 0);
            __builtin_amdgcn_global_load_async_to_lds_b128(g1, l3, 16, 0);
            __builtin_amdgcn_global_load_async_to_lds_b128(g1, l3, 32, 0);
            __builtin_amdgcn_global_load_async_to_lds_b128(g1, l3, 48, 0);
            __builtin_amdgcn_s_wait_asynccnt(0);
#else
            const float4* gin = (const float4*)(in + (size_t)row0 * HDIM);
            float4* st = (float4*)Ztf;
            for (int i = tid; i < 32 * HDIM / 4; i += 256) st[i] = gin[i];
#endif
        } else {
            for (int i = tid; i < 32 * HDIM; i += 256) {
                const int grow = row0 + (i >> 7);
                Ztf[i] = (grow < nrows) ? in[(size_t)grow * HDIM + (i & 127)]
                                        : 0.f;
            }
        }
        // prefetch next macro tile while this one converts/computes
        if (row0 + 32 < nrows) {
            __builtin_prefetch(in + (size_t)(row0 + 32) * HDIM + tid * 16, 0, 1);
        }
        __syncthreads();

        // ---- convert to padded bf16 tile, fusing optional BN + ReLU -------
        for (int i = tid; i < 32 * HDIM; i += 256) {
            const int cc = i & 127;
            float v = Ztf[i];
            if (doScale) v = fmaxf(v * scaleIn[cc] + shiftIn[cc], 0.f);
            Zt[(i >> 7) * ZT_STRIDE + cc] = (__bf16)v;
        }
        __syncthreads();

        // ---- preload all K-step fragments so WMMAs issue back-to-back -----
        v16bf a[4], f0[4], f1[4];
#pragma unroll
        for (int ks = 0; ks < 4; ++ks) {
            const int k0 = ks * 32;
#pragma unroll
            for (int e = 0; e < 16; ++e) {
                // ISA 16-bit A 16x32 layout: VGPR pairs, half-split by lane>>4
                const int kk = 2 * ((e >> 1) & 3) + 16 * (e >> 3)
                             + 8 * kh + (e & 1);
                a[ks][e]  = Zt[arow * ZT_STRIDE + k0 + kk];
                // B 32x16: lane%16 = N column, elements = K rows (same kk as A)
                f0[ks][e] = Ws[(k0 + kk) * WS_STRIDE + colBase + r];
                f1[ks][e] = Ws[(k0 + kk) * WS_STRIDE + colBase + 16 + r];
            }
        }
        v8f c0 = {0.f, 0.f, 0.f, 0.f, 0.f, 0.f, 0.f, 0.f};
        v8f c1 = {0.f, 0.f, 0.f, 0.f, 0.f, 0.f, 0.f, 0.f};
#pragma unroll
        for (int ks = 0; ks < 4; ++ks) {
            c0 = __builtin_amdgcn_wmma_f32_16x16x32_bf16(
                     false, a[ks], false, f0[ks], (short)0, c0, false, false);
            c1 = __builtin_amdgcn_wmma_f32_16x16x32_bf16(
                     false, a[ks], false, f1[ks], (short)0, c1, false, false);
        }

        // ---- bias + store + per-column stats (C/D: VGPR j -> row j+8*kh) --
        const int tileRow0 = row0 + rt * 16;
        float sA = 0.f, sqA = 0.f, sB = 0.f, sqB = 0.f;
        if (fullTile) {
            // branch-free stores (uniform condition -> scalar branch)
            float* opA = out + (size_t)(tileRow0 + 8 * kh) * HDIM + colA;
            float* opB = out + (size_t)(tileRow0 + 8 * kh) * HDIM + colB;
#pragma unroll
            for (int j = 0; j < 8; ++j) {
                const float vA = c0[j] + bA;
                const float vB = c1[j] + bB;
                opA[(size_t)j * HDIM] = vA;
                opB[(size_t)j * HDIM] = vB;
                sA += vA; sqA += vA * vA;
                sB += vB; sqB += vB * vB;
            }
        } else {
#pragma unroll
            for (int j = 0; j < 8; ++j) {
                const int grow = tileRow0 + j + 8 * kh;
                float vA = c0[j] + bA;
                float vB = c1[j] + bB;
                if (grow < nrows) {
                    out[(size_t)grow * HDIM + colA] = vA;
                    out[(size_t)grow * HDIM + colB] = vB;
                } else { vA = 0.f; vB = 0.f; }
                sA += vA; sqA += vA * vA;
                sB += vB; sqB += vB * vB;
            }
        }
        if (colsum) {
            sA += __shfl_xor(sA, 16, 32);  sqA += __shfl_xor(sqA, 16, 32);
            sB += __shfl_xor(sB, 16, 32);  sqB += __shfl_xor(sqB, 16, 32);
            if (lane < 16) {
                atomAddF(&colsum[colA], sA);  atomAddF(&colsumsq[colA], sqA);
                atomAddF(&colsum[colB], sB);  atomAddF(&colsumsq[colB], sqB);
            }
        }
    }
}

// ---------------------------------------------------------------------------
// BN parameter fold: scale = g*rsqrt(var+eps), shift = b - mean*scale
// ---------------------------------------------------------------------------
__global__ void bn_params_kernel(const float* __restrict__ colsum,
                                 const float* __restrict__ colsumsq,
                                 const float* __restrict__ g,
                                 const float* __restrict__ b,
                                 float* __restrict__ scale,
                                 float* __restrict__ shift, float inv_n) {
    const int c = threadIdx.x;
    const float m   = colsum[c] * inv_n;
    const float var = colsumsq[c] * inv_n - m * m;
    const float s   = g[c] * rsqrtf(var + BNEPS);
    scale[c] = s;
    shift[c] = b[c] - m * s;
}

// ---------------------------------------------------------------------------
// Elementwise BN + ReLU in place (inter-layer norm), flat indexing
// ---------------------------------------------------------------------------
__global__ __launch_bounds__(256)
void bn_relu_kernel(float* __restrict__ h,
                    const float* __restrict__ scale,
                    const float* __restrict__ shift) {
    const size_t i = (size_t)blockIdx.x * 256 + threadIdx.x;
    const int c = (int)(i & (HDIM - 1));
    h[i] = fmaxf(h[i] * scale[c] + shift[c], 0.f);
}

// ---------------------------------------------------------------------------
// global_add_pool: hg[batch[n]] += h[n]
// ---------------------------------------------------------------------------
__global__ void pool_kernel(const float* __restrict__ h,
                            const int* __restrict__ batch,
                            float* __restrict__ hg) {
    const int n = blockIdx.x;
    const int c = threadIdx.x;
    atomAddF(&hg[(size_t)batch[n] * HDIM + c], h[(size_t)n * HDIM + c]);
}

// ---------------------------------------------------------------------------
// Head: out[g] = relu(hg2[g]) . w_out + b_out   (one wave per graph)
// ---------------------------------------------------------------------------
__global__ void head_kernel(const float* __restrict__ hg2,
                            const float* __restrict__ wout,
                            const float* __restrict__ bout,
                            float* __restrict__ out) {
    const int g = blockIdx.x;
    const int lane = threadIdx.x;
    float s = 0.f;
#pragma unroll
    for (int j = 0; j < 4; ++j) {
        const int c = lane + j * 32;
        s += fmaxf(hg2[(size_t)g * HDIM + c], 0.f) * wout[c];
    }
    s += __shfl_down(s, 16, 32);
    s += __shfl_down(s, 8, 32);
    s += __shfl_down(s, 4, 32);
    s += __shfl_down(s, 2, 32);
    s += __shfl_down(s, 1, 32);
    if (lane == 0) out[g] = s + bout[0];
}

// ---------------------------------------------------------------------------
extern "C" void kernel_launch(void* const* d_in, const int* in_sizes, int n_in,
                              void* d_out, int out_size, void* d_ws, size_t ws_size,
                              hipStream_t stream) {
    (void)in_sizes; (void)n_in; (void)out_size; (void)ws_size;

    const int*   x        = (const int*)d_in[0];
    const int*   ei       = (const int*)d_in[1];
    const int*   batch    = (const int*)d_in[2];
    const float* atom_emb = (const float*)d_in[3];
    const float* conv_w1  = (const float*)d_in[4];
    const float* conv_b1  = (const float*)d_in[5];
    const float* mlp_bn_g = (const float*)d_in[6];
    const float* mlp_bn_b = (const float*)d_in[7];
    const float* conv_w2  = (const float*)d_in[8];
    const float* conv_b2  = (const float*)d_in[9];
    const float* bn_g     = (const float*)d_in[10];
    const float* bn_b     = (const float*)d_in[11];
    const float* lin_h_w  = (const float*)d_in[12];
    const float* lin_h_b  = (const float*)d_in[13];
    const float* lin_o_w  = (const float*)d_in[14];
    const float* lin_o_b  = (const float*)d_in[15];
    float* outp = (float*)d_out;

    const size_t NH = (size_t)NNODES * HDIM;
    const size_t GH = (size_t)NGRAPH * HDIM;
    float* ws   = (float*)d_ws;
    float* h    = ws;                 // 51.2 MB
    float* z    = ws + NH;            // 51.2 MB
    float* t    = ws + 2 * NH;        // 51.2 MB
    float* hg   = ws + 3 * NH;        // 512 KB
    float* hg2  = hg + GH;            // 512 KB
    float* colsum   = hg2 + GH;       // 128
    float* colsumsq = colsum + HDIM;  // 128
    float* scale    = colsumsq + HDIM;
    float* shift    = scale + HDIM;

    const int HH = HDIM * HDIM;
    const int numMacros = NNODES / 32;   // 3125 macro-tiles of 32 rows
    const int macrosPB  = 5;             // 625 blocks
    const float invN = 1.f / (float)NNODES;

    // AtomEncoder
    atom_encoder_kernel<<<NNODES, HDIM, 0, stream>>>(x, atom_emb, h);

    for (int l = 0; l < NLAYERS; ++l) {
        // z = h + segment_sum(h[src], dst)
        (void)hipMemcpyAsync(z, h, NH * sizeof(float),
                             hipMemcpyDeviceToDevice, stream);
        edge_scatter_kernel<<<NEDGES / 8, 256, 0, stream>>>(ei, h, z);

        // t = z @ W1 + b1 (collect BN stats)
        (void)hipMemsetAsync(colsum, 0, 2 * HDIM * sizeof(float), stream);
        gemm_bn_kernel<<<numMacros / macrosPB, 256, 0, stream>>>(
            z, conv_w1 + l * HH, conv_b1 + l * HDIM, nullptr, nullptr,
            t, colsum, colsumsq, NNODES, macrosPB, numMacros);
        bn_params_kernel<<<1, HDIM, 0, stream>>>(
            colsum, colsumsq, mlp_bn_g + l * HDIM, mlp_bn_b + l * HDIM,
            scale, shift, invN);

        // h = relu(BN(t)) @ W2 + b2  (BN+ReLU fused into GEMM input)
        const bool last = (l == NLAYERS - 1);
        if (!last)
            (void)hipMemsetAsync(colsum, 0, 2 * HDIM * sizeof(float), stream);
        gemm_bn_kernel<<<numMacros / macrosPB, 256, 0, stream>>>(
            t, conv_w2 + l * HH, conv_b2 + l * HDIM, scale, shift,
            h, last ? nullptr : colsum, last ? nullptr : colsumsq,
            NNODES, macrosPB, numMacros);

        if (!last) {
            bn_params_kernel<<<1, HDIM, 0, stream>>>(
                colsum, colsumsq, bn_g + l * HDIM, bn_b + l * HDIM,
                scale, shift, invN);
            bn_relu_kernel<<<(int)(NH / 256), 256, 0, stream>>>(h, scale, shift);
        }
    }

    // global_add_pool
    (void)hipMemsetAsync(hg, 0, GH * sizeof(float), stream);
    pool_kernel<<<NNODES, HDIM, 0, stream>>>(h, batch, hg);

    // hg2(pre-relu) = hg @ lin_h_w + b   (32 macro-tiles cover 1000 graphs)
    gemm_bn_kernel<<<32, 256, 0, stream>>>(
        hg, lin_h_w, lin_h_b, nullptr, nullptr,
        hg2, nullptr, nullptr, NGRAPH, 1, 32);

    // out = relu(hg2) @ lin_out_w + b_out
    head_kernel<<<NGRAPH, 32, 0, stream>>>(hg2, lin_o_w, lin_o_b, outp);
}
